// DNA_32916629356554
// MI455X (gfx1250) — matrix operand order repaired and checked by reading, not compiled, not run on hardware
//
#include <hip/hip_runtime.h>
#include <hip/hip_bf16.h>

#define DV 32000
#define DD 1024
#define DE 8
#define DH 4096
#define DT 2048
#define N_HOPS 4
#define RMS_EPS 1e-5f

typedef __attribute__((ext_vector_type(16))) __bf16 v16bf;
typedef __attribute__((ext_vector_type(8)))  float  v8f;

union FragU { uint4 q[2]; v16bf v; };

// A fragment: 16x32 (MxK) bf16, per CDNA5 wave32 layout.
// lanes 0-15: row M=m0+lane, K = {k0..k0+7, k0+16..k0+23}
// lanes 16-31: row M=m0+lane-16, K = {k0+8..k0+15, k0+24..k0+31}
__device__ __forceinline__ v16bf load_a_frag(const __bf16* __restrict__ A, int lda,
                                             int m0, int k0, int lane) {
  const int half = lane >> 4, lr = lane & 15;
  const __bf16* p = A + (size_t)(m0 + lr) * lda + (k0 + half * 8);
  FragU f;
  f.q[0] = *(const uint4*)(p);        // 8 bf16: K base..base+7
  f.q[1] = *(const uint4*)(p + 16);   // 8 bf16: K base+16..base+23
  return f.v;
}

// B fragment: 32x16 (KxN) bf16 from BT[N,K] row-major.
// lanes 0-15: col N=n0+lane, K = {k0..k0+7, k0+16..k0+23}
// lanes 16-31: col N=n0+lane-16, K = {k0+8..k0+15, k0+24..k0+31}
__device__ __forceinline__ v16bf load_b_frag(const __bf16* __restrict__ BT, int ldb,
                                             int n0, int k0, int lane) {
  const int half = lane >> 4, lr = lane & 15;
  const __bf16* p = BT + (size_t)(n0 + lr) * ldb + (k0 + half * 8);
  FragU f;
  f.q[0] = *(const uint4*)(p);
  f.q[1] = *(const uint4*)(p + 16);
  return f.v;
}

// Branch-free tanh-gelu: tanh(y) = 1 - 2/(1+exp(2y)); __expf -> v_exp_f32.
__device__ __forceinline__ float gelu_fast(float x) {
  const float c0 = 0.7978845608028654f;   // sqrt(2/pi)
  const float c1 = 0.044715f;
  const float y = c0 * (x + c1 * x * x * x);
  const float t = 1.0f - 2.0f / (1.0f + __expf(2.0f * y));
  return 0.5f * x * (1.0f + t);
}

// MODE 0: Cb(bf16) = gelu(A @ B)            (expert GEMM1 -> u)
// MODE 1: C(f32) (+)= probs[row,eidx]*(A@B) (expert GEMM2 -> delta; first overwrites)
// MODE 2: C(f32) = A @ B                    (final logits)
// Block tile: 128(M) x 256(N); 8 waves as 2(M) x 4(N); wave tile 64x64 (4x4 WMMA).
template <int MODE>
__global__ __launch_bounds__(256) void k_gemm(
    const __bf16* __restrict__ A,    // [M,K] bf16 row-major
    const __bf16* __restrict__ BT,   // [N,K] bf16 row-major (B transposed)
    float* __restrict__ C,
    __bf16* __restrict__ Cb,
    const float* __restrict__ probs, int eidx, int first,
    int N, int K)
{
  const int lane = threadIdx.x & 31;
  const int wave = threadIdx.x >> 5;                       // 8 waves
  const int mBase = blockIdx.y * 128 + (wave & 1) * 64;    // 2 waves along M
  const int nBase = blockIdx.x * 256 + (wave >> 1) * 64;   // 4 waves along N

  v8f acc[4][4] = {};   // 64x64 wave tile

  for (int k0 = 0; k0 < K; k0 += 32) {
    if (k0 + 32 < K) {
      __builtin_prefetch(A  + (size_t)(mBase + (lane & 15)) * K + k0 + 32, 0, 3);
      __builtin_prefetch(BT + (size_t)(nBase + (lane & 15)) * K + k0 + 32, 0, 3);
    }
    v16bf a[4], b[4];
#pragma unroll
    for (int i = 0; i < 4; ++i) a[i] = load_a_frag(A, K, mBase + 16 * i, k0, lane);
#pragma unroll
    for (int j = 0; j < 4; ++j) b[j] = load_b_frag(BT, K, nBase + 16 * j, k0, lane);
#pragma unroll
    for (int j = 0; j < 4; ++j)
#pragma unroll
      for (int i = 0; i < 4; ++i)
        acc[i][j] = __builtin_amdgcn_wmma_f32_16x16x32_bf16(
            false, a[i], false, b[j], (short)0, acc[i][j], false, false);
  }

  // C/D layout: VGPR r <-> M = r (lanes 0-15) or r+8 (lanes 16-31); N = lane&15
  const int half = lane >> 4, lr = lane & 15;
#pragma unroll
  for (int i = 0; i < 4; ++i) {
#pragma unroll
    for (int j = 0; j < 4; ++j) {
      const int n0 = nBase + 16 * j;
#pragma unroll
      for (int r = 0; r < 8; ++r) {
        const int row = mBase + 16 * i + half * 8 + r;
        const size_t o = (size_t)row * N + n0 + lr;
        const float v = acc[i][j][r];
        if (MODE == 0) {
          Cb[o] = (__bf16)gelu_fast(v);
        } else if (MODE == 1) {
          const float p = probs[row * DE + eidx];
          C[o] = first ? (p * v) : (C[o] + p * v);
        } else {
          C[o] = v;
        }
      }
    }
  }
}

__global__ __launch_bounds__(256) void k_gather_embed(const int* __restrict__ ids,
                                                      const float* __restrict__ embedW,
                                                      float* __restrict__ h) {
  const int i = blockIdx.x * 256 + threadIdx.x;   // i < T*D
  const int t = i >> 10, d = i & 1023;
  h[i] = embedW[(size_t)ids[t] * DD + d];
}

__global__ __launch_bounds__(256) void k_cast_bf16(const float* __restrict__ in,
                                                   __bf16* __restrict__ out, int n) {
  const int i = blockIdx.x * 256 + threadIdx.x;
  if (i < n) out[i] = (__bf16)in[i];
}

// in: batched [K,N] f32 ; out: batched [N,K] bf16
__global__ __launch_bounds__(256) void k_transpose_cast(const float* __restrict__ in,
                                                        __bf16* __restrict__ out,
                                                        int K, int N) {
  __shared__ float tile[32][33];
  const float* src = in  + (size_t)blockIdx.z * K * N;
  __bf16*      dst = out + (size_t)blockIdx.z * K * N;
  const int nb = blockIdx.x * 32, kb = blockIdx.y * 32;
  const int tx = threadIdx.x & 31, ty = threadIdx.x >> 5;   // ty: 0..7
#pragma unroll
  for (int r = ty; r < 32; r += 8) tile[r][tx] = src[(size_t)(kb + r) * N + nb + tx];
  __syncthreads();
#pragma unroll
  for (int r = ty; r < 32; r += 8) dst[(size_t)(nb + r) * K + kb + tx] = (__bf16)tile[tx][r];
}

__global__ __launch_bounds__(256) void k_inv_rms(const float* __restrict__ h,
                                                 float* __restrict__ irms) {
  __shared__ float s[256];
  const int t = blockIdx.x;
  float a = 0.f;
  for (int d = threadIdx.x; d < DD; d += 256) {
    const float x = h[(size_t)t * DD + d];
    a += x * x;
  }
  s[threadIdx.x] = a;
  __syncthreads();
  for (int o = 128; o > 0; o >>= 1) {
    if (threadIdx.x < o) s[threadIdx.x] += s[threadIdx.x + o];
    __syncthreads();
  }
  if (threadIdx.x == 0) irms[t] = rsqrtf(s[0] / DD + RMS_EPS);
}

__global__ __launch_bounds__(256) void k_scale_cast(const float* __restrict__ h,
                                                    const float* __restrict__ irms,
                                                    const float* __restrict__ w,
                                                    __bf16* __restrict__ out) {
  const int i = blockIdx.x * 256 + threadIdx.x;   // i < T*D
  const int t = i >> 10, d = i & 1023;
  out[i] = (__bf16)(h[i] * irms[t] * w[d]);
}

__global__ __launch_bounds__(256) void k_add(float* __restrict__ h,
                                             const float* __restrict__ delta) {
  const int i = blockIdx.x * 256 + threadIdx.x;
  h[i] += delta[i];
}

// Per token: logits[e] = (rmsnorm(h)*lnw) . Wr[:,e]; probs = where(top-k, softmax, 0)
__global__ __launch_bounds__(256) void k_router(const float* __restrict__ h,
                                                const float* __restrict__ irms,
                                                const float* __restrict__ lnw,
                                                const float* __restrict__ Wr,  // [D,E]
                                                const int* __restrict__ kptr,
                                                float* __restrict__ probs) {
  __shared__ float s[DE][256];
  const int t = blockIdx.x;
  const float ir = irms[t];
  float part[DE];
#pragma unroll
  for (int e = 0; e < DE; ++e) part[e] = 0.f;
  for (int d = threadIdx.x; d < DD; d += 256) {
    const float x = h[(size_t)t * DD + d] * ir * lnw[d];
#pragma unroll
    for (int e = 0; e < DE; ++e) part[e] += x * Wr[(size_t)d * DE + e];
  }
#pragma unroll
  for (int e = 0; e < DE; ++e) s[e][threadIdx.x] = part[e];
  __syncthreads();
  for (int o = 128; o > 0; o >>= 1) {
    if (threadIdx.x < o)
#pragma unroll
      for (int e = 0; e < DE; ++e) s[e][threadIdx.x] += s[e][threadIdx.x + o];
    __syncthreads();
  }
  if (threadIdx.x == 0) {
    float lg[DE], ex[DE], m = -1e30f, sum = 0.f;
#pragma unroll
    for (int e = 0; e < DE; ++e) { lg[e] = s[e][0]; m = fmaxf(m, lg[e]); }
#pragma unroll
    for (int e = 0; e < DE; ++e) { ex[e] = expf(lg[e] - m); sum += ex[e]; }
    int k = kptr[0]; if (k > DE) k = DE;
    bool keep[DE] = {};
    for (int it = 0; it < k; ++it) {
      int best = -1; float bv = -1e38f;
      for (int e = 0; e < DE; ++e)
        if (!keep[e] && lg[e] > bv) { bv = lg[e]; best = e; }
      if (best >= 0) keep[best] = true;
    }
    const float inv = 1.0f / sum;
#pragma unroll
    for (int e = 0; e < DE; ++e) probs[(size_t)t * DE + e] = keep[e] ? ex[e] * inv : 0.f;
  }
}

extern "C" void kernel_launch(void* const* d_in, const int* in_sizes, int n_in,
                              void* d_out, int out_size, void* d_ws, size_t ws_size,
                              hipStream_t stream) {
  const int*   ids    = (const int*)d_in[0];
  const float* embedW = (const float*)d_in[1];
  const float* rlnw   = (const float*)d_in[2];   // [NHOPS, D]
  const float* rW     = (const float*)d_in[3];   // [NHOPS, D, E]
  const float* elnw   = (const float*)d_in[4];   // [E, D]
  const float* eW1    = (const float*)d_in[5];   // [E, D, H]
  const float* eW2    = (const float*)d_in[6];   // [E, H, D]
  const float* olnw   = (const float*)d_in[7];   // [D]
  const int*   kptr   = (const int*)d_in[8];
  float* out = (float*)d_out;                    // [T, V] f32
  (void)in_sizes; (void)n_in; (void)out_size; (void)ws_size;

  char* ws = (char*)d_ws;
  auto carve = [&](size_t bytes) -> char* {
    char* p = ws;
    ws += (bytes + 255) & ~(size_t)255;
    return p;
  };
  __bf16* w1t   = (__bf16*)carve((size_t)DE * DH * DD * 2);  // [E][H,D] bf16 (W1^T)
  __bf16* w2t   = (__bf16*)carve((size_t)DE * DD * DH * 2);  // [E][D,H] bf16 (W2^T)
  __bf16* embB  = (__bf16*)carve((size_t)DV * DD * 2);       // [V,D] bf16
  float*  h     = (float*)carve((size_t)DT * DD * 4);
  float*  delta = (float*)carve((size_t)DT * DD * 4);
  float*  irms  = (float*)carve((size_t)DT * 4);
  float*  probs = (float*)carve((size_t)DT * DE * 4);
  __bf16* xn    = (__bf16*)carve((size_t)DT * DD * 2);
  __bf16* u     = (__bf16*)carve((size_t)DT * DH * 2);

  // ---- one-time weight prep (bf16 + transpose into BT layout) ----
  k_transpose_cast<<<dim3(DH / 32, DD / 32, DE), 256, 0, stream>>>(eW1, w1t, DD, DH);
  k_transpose_cast<<<dim3(DD / 32, DH / 32, DE), 256, 0, stream>>>(eW2, w2t, DH, DD);
  {
    const int n = DV * DD;
    k_cast_bf16<<<(n + 255) / 256, 256, 0, stream>>>(embedW, embB, n);
  }
  k_gather_embed<<<(DT * DD) / 256, 256, 0, stream>>>(ids, embedW, h);

  // ---- hops ----
  for (int hop = 0; hop < N_HOPS; ++hop) {
    k_inv_rms<<<DT, 256, 0, stream>>>(h, irms);
    k_router<<<DT, 256, 0, stream>>>(h, irms, rlnw + (size_t)hop * DD,
                                     rW + (size_t)hop * DD * DE, kptr, probs);
    for (int e = 0; e < DE; ++e) {
      k_scale_cast<<<(DT * DD) / 256, 256, 0, stream>>>(h, irms, elnw + (size_t)e * DD, xn);
      // u = gelu(xn @ W1[e])    [T,H]
      k_gemm<0><<<dim3(DH / 256, DT / 128), 256, 0, stream>>>(
          xn, w1t + (size_t)e * DH * DD, nullptr, u, nullptr, 0, 0, DH, DD);
      // delta (+)= probs[:,e] * (u @ W2[e])   [T,D]
      k_gemm<1><<<dim3(DD / 256, DT / 128), 256, 0, stream>>>(
          u, w2t + (size_t)e * DD * DH, delta, nullptr, probs, e, (e == 0) ? 1 : 0, DD, DH);
    }
    k_add<<<(DT * DD) / 256, 256, 0, stream>>>(h, delta);
  }

  // ---- final tied-embedding logits ----
  k_inv_rms<<<DT, 256, 0, stream>>>(h, irms);
  k_scale_cast<<<(DT * DD) / 256, 256, 0, stream>>>(h, irms, olnw, xn);
  k_gemm<2><<<dim3(DV / 256, DT / 128), 256, 0, stream>>>(
      xn, embB, out, nullptr, nullptr, 0, 0, DV, DD);
}
